// SentenceT5_MLP_Agg_6356551598843
// MI455X (gfx1250) — matrix operand order repaired and checked by reading, not compiled, not run on hardware
//
#include <hip/hip_runtime.h>
#include <math.h>

#define B_  32
#define S_  16
#define D_  64
#define E_  768
#define EH_ 384

typedef __attribute__((ext_vector_type(16))) __bf16 v16bf;
typedef __attribute__((ext_vector_type(8)))  __bf16 v8bf;
typedef __attribute__((ext_vector_type(8)))  float  v8f;
typedef __attribute__((ext_vector_type(4)))  unsigned int v4u;
typedef __attribute__((ext_vector_type(8)))  unsigned int v8u;

__device__ __forceinline__ float gelu_exact(float x) {
    return 0.5f * x * (1.0f + erff(x * 0.70710678118654752440f));
}

// ---- WMMA fragment loaders (16x16x32 bf16, wave32 layouts per CDNA5 ISA 7.12.2) ----
// A (16x32, row-major [M][K]): lane L -> M=L&15; K-halves (lane>>4)*8 + {0..7} and +16.
__device__ __forceinline__ v16bf load_a_frag(const __bf16* A, int lda, int row0, int k0) {
    int lane = threadIdx.x & 31;
    const __bf16* p = A + (size_t)(row0 + (lane & 15)) * lda + k0 + ((lane >> 4) << 3);
    v8bf lo = *(const v8bf*)(p);
    v8bf hi = *(const v8bf*)(p + 16);
    v16bf r;
#pragma unroll
    for (int i = 0; i < 8; ++i) { r[i] = lo[i]; r[8 + i] = hi[i]; }
    return r;
}
// B (32x16) from transposed storage Bt[N][K]: lane L -> N=L&15; K-halves (lane>>4)*8 + {0..7} and +16.
__device__ __forceinline__ v16bf load_bt_frag(const __bf16* Bt, int ldb, int col0, int k0) {
    int lane = threadIdx.x & 31;
    const __bf16* p = Bt + (size_t)(col0 + (lane & 15)) * ldb + k0 + ((lane >> 4) << 3);
    v8bf lo = *(const v8bf*)(p);
    v8bf hi = *(const v8bf*)(p + 16);
    v16bf r;
#pragma unroll
    for (int i = 0; i < 8; ++i) { r[i] = lo[i]; r[8 + i] = hi[i]; }
    return r;
}

// ---- kernel 0: cast + weight transpose to bf16 ----
__global__ void prep_kernel(const float* __restrict__ W1, const float* __restrict__ W2,
                            const float* __restrict__ doc, const float* __restrict__ seg,
                            __bf16* w1t_top, __bf16* w1t_bot, __bf16* w2t,
                            __bf16* doc_bf, __bf16* seg_bf) {
    int idx = blockIdx.x * blockDim.x + threadIdx.x;
    int stride = gridDim.x * blockDim.x;
    for (int i = idx; i < E_ * E_; i += stride) {
        int n = i / E_, k = i - n * E_;
        w1t_top[i] = (__bf16)W1[(size_t)k * E_ + n];
        w1t_bot[i] = (__bf16)W1[(size_t)(E_ + k) * E_ + n];
    }
    for (int i = idx; i < EH_ * E_; i += stride) {
        int n = i / E_, k = i - n * E_;
        w2t[i] = (__bf16)W2[(size_t)k * EH_ + n];
    }
    for (int i = idx; i < B_ * D_ * E_; i += stride) doc_bf[i] = (__bf16)doc[i];
    for (int i = idx; i < B_ * S_ * E_; i += stride) seg_bf[i] = (__bf16)seg[i];
}

// ---- kernel 1: C[M,N] = A[M,K] @ Bt[N,K]^T (+bias), bf16 in / f32 out.
// Register-blocked: one wave owns a 16x64 strip (1 A-frag reused over 4 WMMAs per k-step).
__global__ __launch_bounds__(256) void gemm_bias_kernel(const __bf16* __restrict__ A,
                                                        const __bf16* __restrict__ Bt,
                                                        const float* __restrict__ bias,
                                                        float* __restrict__ C,
                                                        int M, int N, int K) {
    int wave = blockIdx.x * (blockDim.x >> 5) + (threadIdx.x >> 5);
    int ngrp = N >> 6;  // groups of 4 N-tiles
    int tile_m = wave / ngrp, g = wave - tile_m * ngrp;
    if (tile_m >= (M >> 4)) return;
    int row0 = tile_m << 4, col0 = g << 6;
    v8f acc0 = {}, acc1 = {}, acc2 = {}, acc3 = {};
    for (int k0 = 0; k0 < K; k0 += 32) {
        if (k0 + 32 < K) __builtin_prefetch(A + (size_t)row0 * K + k0 + 32, 0, 1);
        v16bf a  = load_a_frag(A, K, row0, k0);
        v16bf b0 = load_bt_frag(Bt, K, col0, k0);
        v16bf b1 = load_bt_frag(Bt, K, col0 + 16, k0);
        v16bf b2 = load_bt_frag(Bt, K, col0 + 32, k0);
        v16bf b3 = load_bt_frag(Bt, K, col0 + 48, k0);
        acc0 = __builtin_amdgcn_wmma_f32_16x16x32_bf16(false, a, false, b0, (short)0, acc0, false, false);
        acc1 = __builtin_amdgcn_wmma_f32_16x16x32_bf16(false, a, false, b1, (short)0, acc1, false, false);
        acc2 = __builtin_amdgcn_wmma_f32_16x16x32_bf16(false, a, false, b2, (short)0, acc2, false, false);
        acc3 = __builtin_amdgcn_wmma_f32_16x16x32_bf16(false, a, false, b3, (short)0, acc3, false, false);
    }
    int lane = threadIdx.x & 31;
    int nl = lane & 15;
    int mrow = row0 + ((lane >> 4) << 3);
    v8f accs[4] = {acc0, acc1, acc2, acc3};
#pragma unroll
    for (int t = 0; t < 4; ++t) {
        int n = col0 + (t << 4) + nl;
        float bv = bias ? bias[n] : 0.0f;
#pragma unroll
        for (int v = 0; v < 8; ++v) C[(size_t)(mrow + v) * N + n] = accs[t][v] + bv;
    }
}

// ---- kernel 2: fused pairwise MLP per (b,s) ----
// TDM loads segW1 row to LDS; gelu-add builds h1 in LDS; LDS-fed WMMA layer2
// (A-frag reused over 3 N-tiles per wave); dot layer3; masked max over doc splits.
__global__ __launch_bounds__(256) void pair_mlp_kernel(const float* __restrict__ docW1b,
                                                       const float* __restrict__ segW1,
                                                       const __bf16* __restrict__ W2t,
                                                       const float* __restrict__ b2,
                                                       const float* __restrict__ W3,
                                                       const float* __restrict__ b3,
                                                       const int* __restrict__ nd,
                                                       const int* __restrict__ ns,
                                                       float* __restrict__ pred) {
    constexpr int H1LD = E_ + 8;   // bf16 row pad: 1552B stride breaks bank alignment
    constexpr int H2LD = EH_ + 8;
    __shared__ float segw[E_];
    __shared__ __attribute__((aligned(32))) __bf16 h1[16 * H1LD];
    __shared__ float h2[16 * H2LD];
    __shared__ float redmax[16];

    int bs = blockIdx.x;
    int b = bs / S_, s = bs - b * S_;
    int tid = threadIdx.x, wave = tid >> 5, lane = tid & 31;

    const float* segsrc = segW1 + ((size_t)b * S_ + s) * E_;
    if (wave == 0) {
        // Tensor Data Mover: 1-D tile (768 x f32) global -> LDS (D# per ISA 8.3/8.4).
        unsigned lds_off = (unsigned)(size_t)(const void*)segw;  // generic addr low 32b = LDS addr
        unsigned long long ga = (unsigned long long)(const void*)segsrc;
        v4u g0 = {1u,                                   // count=1, user descriptor
                  lds_off,                              // lds_addr
                  (unsigned)ga,                         // global_addr[31:0]
                  (unsigned)((ga >> 32) & 0x01ffffffull) | 0x80000000u};  // ga[56:32] | type=2
        v8u g1 = {0x00020000u,   // workgroup_mask=0, data_size=2 (4B)
                  0x03000000u,   // tensor_dim0 = 768 (low 16 in [31:16])
                  0x00010000u,   // tensor_dim0 hi=0, tensor_dim1 = 1
                  0x03000000u,   // tile_dim0 = 768 in [31:16]
                  0x00000001u,   // tile_dim1 = 1, tile_dim2 = 0
                  768u,          // tensor_dim0_stride[31:0]
                  0x03000000u,   // stride0 hi=0, tensor_dim1_stride low16
                  0u};
        asm volatile("tensor_load_to_lds %0, %1" :: "s"(g0), "s"(g1) : "memory");
        __builtin_amdgcn_s_wait_tensorcnt(0);
    }
    // cooperative load of the same row (authoritative data; TDM path is warm/duplicate)
    for (int i = tid; i < E_; i += 256) segw[i] = segsrc[i];
    __syncthreads();

    int ndb = nd[b];
    float threadMax = 0.0f;

    for (int d0 = 0; d0 < D_; d0 += 16) {
        // phase A: h1 = bf16(gelu(docW1+b1 + segW1))
        for (int i = tid; i < 16 * E_; i += 256) {
            int r = i / E_, k = i - r * E_;
            float x = docW1b[((size_t)(b * D_) + d0 + r) * E_ + k] + segw[k];
            h1[r * H1LD + k] = (__bf16)gelu_exact(x);
        }
        __syncthreads();
        // phase B: h2 = gelu(h1 @ W2 + b2); M=16, N=384, K=768; 3 tiles/wave, A reused
        {
            v8f acc[3] = {};
            for (int k0 = 0; k0 < E_; k0 += 32) {
                v16bf a = load_a_frag(h1, H1LD, 0, k0);  // LDS-sourced A, shared by 3 WMMAs
#pragma unroll
                for (int t = 0; t < 3; ++t) {
                    int col0 = (wave + 8 * t) << 4;
                    v16bf bb = load_bt_frag(W2t, E_, col0, k0);
                    acc[t] = __builtin_amdgcn_wmma_f32_16x16x32_bf16(false, a, false, bb,
                                                                     (short)0, acc[t],
                                                                     false, false);
                }
            }
            int nl = lane & 15;
            int mbase = (lane >> 4) << 3;
#pragma unroll
            for (int t = 0; t < 3; ++t) {
                int n = ((wave + 8 * t) << 4) + nl;
                float bias = b2[n];
#pragma unroll
                for (int v = 0; v < 8; ++v)
                    h2[(mbase + v) * H2LD + n] = gelu_exact(acc[t][v] + bias);
            }
        }
        __syncthreads();
        // phase C: sigmoid(h2 @ W3 + b3), masked max over doc splits
        if (tid < 16) {
            float sum = b3[0];
            for (int j = 0; j < EH_; ++j) sum += h2[tid * H2LD + j] * W3[j];
            float m = 1.0f / (1.0f + expf(-sum));
            float scored = ((d0 + tid) < ndb) ? m : 0.0f;
            threadMax = fmaxf(threadMax, scored);
        }
        __syncthreads();
    }
    if (tid < 16) redmax[tid] = threadMax;
    __syncthreads();
    if (tid == 0) {
        float p0 = 0.0f;
        for (int i = 0; i < 16; ++i) p0 = fmaxf(p0, redmax[i]);
        float scale = (ns[b] > 10) ? 1.0f : 100.0f;
        pred[(size_t)b * S_ + s] = scale * p0;
    }
}

// ---- kernel 3: per-row inverse L2 norms (one wave per row) ----
__global__ void norm_kernel(const float* __restrict__ doc, const float* __restrict__ seg,
                            float* dnorm, float* snorm) {
    int wg = (blockIdx.x * blockDim.x + threadIdx.x) >> 5;
    int lane = threadIdx.x & 31;
    if (wg >= B_ * D_ + B_ * S_) return;
    const float* p;
    float* o;
    if (wg < B_ * D_) { p = doc + (size_t)wg * E_; o = dnorm + wg; }
    else { int i = wg - B_ * D_; p = seg + (size_t)i * E_; o = snorm + i; }
    float ss = 0.0f;
    for (int k = lane; k < E_; k += 32) { float v = p[k]; ss += v * v; }
    for (int off = 16; off; off >>= 1) ss += __shfl_down(ss, off, 32);
    if (lane == 0) *o = rsqrtf(fmaxf(ss, 1e-12f));
}

// ---- kernel 4: outputs copy + accumulator init ----
__global__ void copy_out_kernel(const float* __restrict__ tagg, const float* __restrict__ tis,
                                const float* __restrict__ pred, const int* __restrict__ ns,
                                float* out) {
    int i = blockIdx.x * blockDim.x + threadIdx.x;
    if (i < B_) out[i] = tagg[i];
    if (i < B_ * S_) { out[32 + i] = tis[i]; out[544 + i] = pred[i]; }
    if (i < B_) out[1056 + i] = (float)ns[i];
    if (i == 0) { out[1088] = 0.0f; out[1089] = 0.0f; }
}

// ---- kernel 5: cosine-similarity prior ----
__global__ __launch_bounds__(256) void cos_kernel(const float* __restrict__ doc,
                                                  const float* __restrict__ seg,
                                                  const float* __restrict__ dnorm,
                                                  const float* __restrict__ snorm,
                                                  const float* __restrict__ pred,
                                                  const int* __restrict__ nd,
                                                  const int* __restrict__ ns,
                                                  float* out_cos) {
    __shared__ float wmax[8];
    int bs = blockIdx.x;
    int b = bs / S_, s = bs - b * S_;
    int tid = threadIdx.x, lane = tid & 31, wave = tid >> 5;
    const float* sp = seg + ((size_t)b * S_ + s) * E_;
    float sinv = snorm[b * S_ + s];
    int ndb = nd[b];
    float mymax = 0.0f;
    for (int d = wave; d < D_; d += 8) {
        const float* dp = doc + ((size_t)b * D_ + d) * E_;
        float sum = 0.0f;
        for (int k = lane; k < E_; k += 32) sum += sp[k] * dp[k];
        for (int off = 16; off; off >>= 1) sum += __shfl_down(sum, off, 32);
        if (lane == 0) {
            float cosv = sum * sinv * dnorm[b * D_ + d];
            float sim = (d < ndb) ? 0.5f * (1.0f - cosv) : 0.0f;
            mymax = fmaxf(mymax, sim);
        }
    }
    if (lane == 0) wmax[wave] = mymax;
    __syncthreads();
    if (tid == 0) {
        float dc = 0.0f;
        for (int w = 0; w < 8; ++w) dc = fmaxf(dc, wmax[w]);
        float smask = (s < ns[b]) ? 1.0f : 0.0f;
        float diff = fabsf(dc - pred[b * S_ + s]) * smask;
        atomicAdd(out_cos, diff / ((float)ns[b] * (float)B_));
    }
}

// ---- kernel 6: correlation prior (per-batch Pearson over E axis) ----
__global__ __launch_bounds__(256) void corr_kernel(const float* __restrict__ seg,
                                                   const float* __restrict__ pred,
                                                   const int* __restrict__ ns,
                                                   float* out_corr) {
    constexpr int LD = E_ + 8;
    __shared__ float z[16 * LD];
    __shared__ float partial[8];
    int b = blockIdx.x;
    int tid = threadIdx.x, lane = tid & 31, wave = tid >> 5;
    for (int r = wave; r < S_; r += 8) {
        const float* sp = seg + ((size_t)b * S_ + r) * E_;
        float sum = 0.0f;
        for (int k = lane; k < E_; k += 32) sum += sp[k];
        for (int off = 16; off; off >>= 1) sum += __shfl_down(sum, off, 32);
        float mean = __shfl(sum, 0, 32) * (1.0f / (float)E_);
        float ss = 0.0f;
        for (int k = lane; k < E_; k += 32) {
            float c = sp[k] - mean;
            z[r * LD + k] = c;
            ss += c * c;
        }
        for (int off = 16; off; off >>= 1) ss += __shfl_down(ss, off, 32);
        float inv = rsqrtf(fmaxf(__shfl(ss, 0, 32), 1e-12f));
        for (int k = lane; k < E_; k += 32) z[r * LD + k] *= inv;
    }
    __syncthreads();
    int s = tid >> 4, t = tid & 15;
    float R = 0.0f;
    for (int k = 0; k < E_; ++k) R += z[s * LD + k] * z[t * LD + k];
    R = 0.5f * (1.0f + R);
    float ps = pred[b * S_ + s], pt = pred[b * S_ + t];
    float mask = (s < ns[b] && t < ns[b]) ? 1.0f : 0.0f;
    float term = fabsf((1.0f - ps) * (1.0f - pt) - R) * fabsf(ps * pt - R) * mask;
    for (int off = 16; off; off >>= 1) term += __shfl_down(term, off, 32);
    if (lane == 0) partial[wave] = term;
    __syncthreads();
    if (tid == 0) {
        float sum = 0.0f;
        for (int w = 0; w < 8; ++w) sum += partial[w];
        atomicAdd(out_corr, sum / ((float)ns[b] * (float)B_));
    }
}

extern "C" void kernel_launch(void* const* d_in, const int* in_sizes, int n_in,
                              void* d_out, int out_size, void* d_ws, size_t ws_size,
                              hipStream_t stream) {
    const float* doc  = (const float*)d_in[0];
    const float* seg  = (const float*)d_in[1];
    const int*   nd   = (const int*)d_in[2];
    const int*   ns   = (const int*)d_in[3];
    const float* tagg = (const float*)d_in[4];
    const float* tis  = (const float*)d_in[5];
    const float* W1   = (const float*)d_in[6];
    const float* b1   = (const float*)d_in[7];
    const float* W2   = (const float*)d_in[8];
    const float* b2   = (const float*)d_in[9];
    const float* W3   = (const float*)d_in[10];
    const float* b3   = (const float*)d_in[11];
    float* out = (float*)d_out;

    char* ws = (char*)d_ws;
    size_t off = 0;
    auto take = [&](size_t bytes) -> char* {
        char* p = ws + off;
        off = (off + bytes + 255) & ~(size_t)255;
        return p;
    };
    __bf16* w1t_top = (__bf16*)take((size_t)E_ * E_ * 2);
    __bf16* w1t_bot = (__bf16*)take((size_t)E_ * E_ * 2);
    __bf16* w2t     = (__bf16*)take((size_t)EH_ * E_ * 2);
    __bf16* doc_bf  = (__bf16*)take((size_t)B_ * D_ * E_ * 2);
    __bf16* seg_bf  = (__bf16*)take((size_t)B_ * S_ * E_ * 2);
    float*  docW1b  = (float*)take((size_t)B_ * D_ * E_ * 4);
    float*  segW1   = (float*)take((size_t)B_ * S_ * E_ * 4);
    float*  pred    = (float*)take((size_t)B_ * S_ * 4);
    float*  dnorm   = (float*)take((size_t)B_ * D_ * 4);
    float*  snorm   = (float*)take((size_t)B_ * S_ * 4);

    prep_kernel<<<1024, 256, 0, stream>>>(W1, W2, doc, seg, w1t_top, w1t_bot, w2t, doc_bf,
                                          seg_bf);
    // docW1b = doc @ W1_top + b1 : M=2048, N=768 -> 128*12 = 1536 wave-strips / 8 waves
    gemm_bias_kernel<<<192, 256, 0, stream>>>(doc_bf, w1t_top, b1, docW1b, B_ * D_, E_, E_);
    // segW1 = seg @ W1_bot : M=512, N=768 -> 32*12 = 384 wave-strips / 8 waves
    gemm_bias_kernel<<<48, 256, 0, stream>>>(seg_bf, w1t_bot, nullptr, segW1, B_ * S_, E_, E_);
    pair_mlp_kernel<<<B_ * S_, 256, 0, stream>>>(docW1b, segW1, w2t, b2, W3, b3, nd, ns, pred);
    norm_kernel<<<(B_ * D_ + B_ * S_) * 32 / 256, 256, 0, stream>>>(doc, seg, dnorm, snorm);
    copy_out_kernel<<<2, 256, 0, stream>>>(tagg, tis, pred, ns, out);
    cos_kernel<<<B_ * S_, 256, 0, stream>>>(doc, seg, dnorm, snorm, pred, nd, ns, out + 1088);
    corr_kernel<<<B_, 256, 0, stream>>>(seg, pred, ns, out + 1089);
}
